// MultiHeadAttention_11433202942645
// MI455X (gfx1250) — matrix operand order repaired and checked by reading, not compiled, not run on hardware
//
#include <hip/hip_runtime.h>

// ---------------------------------------------------------------------------
// MultiHeadAttention for MI455X (gfx1250), wave32 + v_wmma_f32_16x16x32_bf16,
// async global->LDS staging (ASYNCcnt) for the attention K/V stream.
// B=2, S=2048, D=1024, H=16, DK=64
// ---------------------------------------------------------------------------

#define S_LEN 2048
#define D_DIM 1024
#define H_NUM 16
#define DK_SZ 64

typedef __bf16 v16bf __attribute__((ext_vector_type(16)));
typedef float  v8f   __attribute__((ext_vector_type(8)));
typedef unsigned int v4u __attribute__((ext_vector_type(4)));

union FragBF {
  v16bf v;
  v4u   q[2];
  unsigned short h[16];
};

// fp32 -> bf16, round-to-nearest-even
__device__ __forceinline__ unsigned short f2bf(float f) {
  unsigned int u = __float_as_uint(f);
  u += 0x7FFFu + ((u >> 16) & 1u);
  return (unsigned short)(u >> 16);
}

__device__ __forceinline__ v8f wmma_bf16(const FragBF& a, const FragBF& b, v8f c) {
  return __builtin_amdgcn_wmma_f32_16x16x32_bf16(false, a.v, false, b.v,
                                                 (short)0, c, false, false);
}

// 16-byte global -> LDS copy. Preferred: gfx1250 async-to-LDS (ASYNCcnt, no
// VGPR bounce). Builtin prototype (from hipcc diagnostics):
//   void (...)(int4 addrspace(1)*, int4 addrspace(3)*, imm int off, imm int cpol)
// On amdgcn a generic global address equals its AS(1) address, and the low
// 32 bits of a generic LDS address are the LDS byte offset.
typedef int v4i_vs __attribute__((vector_size(16)));
typedef __attribute__((address_space(1))) v4i_vs* gptr_v4;
typedef __attribute__((address_space(3))) v4i_vs* lptr_v4;

__device__ __forceinline__ void async_copy16(const void* g, void* l) {
#if __has_builtin(__builtin_amdgcn_global_load_async_to_lds_b128)
  __builtin_amdgcn_global_load_async_to_lds_b128(
      (gptr_v4)(uintptr_t)g, (lptr_v4)(unsigned int)(uintptr_t)l, 0, 0);
#else
  *(v4u*)l = *(const v4u*)g;
#endif
}

__device__ __forceinline__ void wait_async_lds() {
  asm volatile("s_wait_asynccnt 0x0" ::: "memory");
}

// ---------------------------------------------------------------------------
// proj_gemm<MODE>: Y = X @ W + bias, tiled 64x64 per WG (4 waves), BK=32.
//   MODE 0/1: fp32 X -> bf16 out, head-split  [b][h][s][dk]     (Q, K)
//   MODE 2  : fp32 X -> bf16 out, transposed  [b][h][dk][s]     (V)
//   MODE 3  : bf16 X -> fp32 out, flat        [m][n]            (final proj)
// ---------------------------------------------------------------------------
template <int MODE>
__global__ __launch_bounds__(128) void proj_gemm(
    const float* __restrict__ Xf, const unsigned short* __restrict__ Xb,
    const float* __restrict__ W, const float* __restrict__ bias,
    unsigned short* __restrict__ outB, float* __restrict__ outF) {
  __shared__ alignas(16) unsigned short As[64][32];  // [m][k] bf16
  __shared__ alignas(16) unsigned short Bs[64][32];  // [n][k] bf16 (W^T tile)

  const int tid  = threadIdx.x;
  const int wave = tid >> 5;
  const int lane = tid & 31;
  const int ln   = lane & 15;
  const int hi   = lane >> 4;
  const int mbase = blockIdx.y * 64;
  const int nbase = blockIdx.x * 64;

  v8f acc[4] = { {}, {}, {}, {} };

  for (int k0 = 0; k0 < D_DIM; k0 += 32) {
    // cooperative tile staging (2048 elems each, 128 threads, 16 iters)
    #pragma unroll
    for (int i = 0; i < 16; ++i) {
      const int idx = tid + i * 128;
      const int m = idx >> 5, kk = idx & 31;
      if (MODE == 3)
        As[m][kk] = Xb[(size_t)(mbase + m) * D_DIM + k0 + kk];
      else
        As[m][kk] = f2bf(Xf[(size_t)(mbase + m) * D_DIM + k0 + kk]);
      const int kw = idx >> 6, n = idx & 63;
      Bs[n][kw] = f2bf(W[(size_t)(k0 + kw) * D_DIM + nbase + n]);
    }
    __syncthreads();

    // A fragment: row M = wave*16 + ln; 16 contiguous K starting at 16*hi
    FragBF a;
    const unsigned short* ar = &As[wave * 16 + ln][0];
    a.q[0] = *(const v4u*)(ar + 16 * hi);
    a.q[1] = *(const v4u*)(ar + 16 * hi + 8);

    #pragma unroll
    for (int nf = 0; nf < 4; ++nf) {
      // B fragment: col N = nf*16 + ln; 16 contiguous K starting at 16*hi
      FragBF bb;
      const unsigned short* br = &Bs[nf * 16 + ln][0];
      bb.q[0] = *(const v4u*)(br + 16 * hi);
      bb.q[1] = *(const v4u*)(br + 16 * hi + 8);
      acc[nf] = wmma_bf16(a, bb, acc[nf]);
    }
    __syncthreads();
  }

  // epilogue: C layout is (M = r + 8*hi, N = ln) per VGPR r
  const int mrow0 = mbase + wave * 16;
  #pragma unroll
  for (int nf = 0; nf < 4; ++nf) {
    const int ncol = nbase + nf * 16 + ln;
    const float bv = bias[ncol];
    if (MODE == 3) {
      #pragma unroll
      for (int r = 0; r < 8; ++r) {
        const int m = mrow0 + r + 8 * hi;
        outF[(size_t)m * D_DIM + ncol] = acc[nf][r] + bv;
      }
    } else if (MODE == 2) {
      // V transposed: [b][h][dk][s]; 8 consecutive s -> one b128 store
      const int h = ncol >> 6, dk = ncol & 63;
      const int m0 = mrow0 + 8 * hi;
      const int bb = m0 >> 11, ss = m0 & (S_LEN - 1);
      union { v4u u; unsigned short h8[8]; } pk;
      #pragma unroll
      for (int r = 0; r < 8; ++r) pk.h8[r] = f2bf(acc[nf][r] + bv);
      *(v4u*)&outB[((size_t)((bb * H_NUM + h) * DK_SZ + dk) << 11) + ss] = pk.u;
    } else {
      // Q/K: [b][h][s][dk]
      const int h = ncol >> 6, dk = ncol & 63;
      #pragma unroll
      for (int r = 0; r < 8; ++r) {
        const int m = mrow0 + r + 8 * hi;
        const int bb = m >> 11, ss = m & (S_LEN - 1);
        outB[((size_t)((bb * H_NUM + h) * S_LEN + ss) << 6) + dk] =
            f2bf(acc[nf][r] + bv);
      }
    }
  }
}

// ---------------------------------------------------------------------------
// flash_attn: causal flash attention.
// 8 waves/WG, 16 query rows per wave (128 q-rows per WG), 32-key tiles.
// K (32x64) and V^T (64x32) tiles are staged into LDS once per WG via
// async-to-LDS (double buffered); each of the 256 threads issues one b128.
// Online softmax with cross-lane (16-group) reductions; P round-trips LDS
// to become a bf16 A-fragment for the P@V WMMA.
// ---------------------------------------------------------------------------
__global__ __launch_bounds__(256) void flash_attn(
    const unsigned short* __restrict__ Qh,
    const unsigned short* __restrict__ Kh,
    const unsigned short* __restrict__ Vt,
    unsigned short* __restrict__ ctx) {
  __shared__ alignas(16) unsigned short Ksh[2][32][64];   // [key][dk]
  __shared__ alignas(16) unsigned short Vsh[2][64][32];   // [dk][key]
  __shared__ alignas(16) unsigned short pbuf[8][16 * 32]; // per-wave P scratch

  const int tid  = threadIdx.x;
  const int wave = tid >> 5;
  const int lane = tid & 31;
  const int ln   = lane & 15;
  const int hi   = lane >> 4;
  const int qB   = blockIdx.x * 128;       // first query row of block
  const int qw   = qB + wave * 16;         // first query row of wave
  const int h    = blockIdx.y;
  const int b    = blockIdx.z;

  const unsigned short* Q = Qh + ((size_t)(b * H_NUM + h) * S_LEN) * DK_SZ;
  const unsigned short* K = Kh + ((size_t)(b * H_NUM + h) * S_LEN) * DK_SZ;
  const unsigned short* V = Vt + ((size_t)(b * H_NUM + h) * DK_SZ) * S_LEN;

  // Q A-fragments (dk 0..31 and 32..63), resident for the whole pass
  FragBF qa0, qa1;
  {
    const unsigned short* qr = Q + (size_t)(qw + ln) * DK_SZ;
    qa0.q[0] = *(const v4u*)(qr + 16 * hi);
    qa0.q[1] = *(const v4u*)(qr + 16 * hi + 8);
    qa1.q[0] = *(const v4u*)(qr + 32 + 16 * hi);
    qa1.q[1] = *(const v4u*)(qr + 32 + 16 * hi + 8);
  }

  float mrow[8], lrow[8], alpha[8];
  v8f o[4] = { {}, {}, {}, {} };
  #pragma unroll
  for (int r = 0; r < 8; ++r) { mrow[r] = -3.0e38f; lrow[r] = 0.0f; }

  // stage helper executed by all 256 threads: one async b128 each per tile
  auto stage = [&](int buf, int kt) {
    {
      const int row = tid >> 3, ch = (tid & 7) * 8;           // K: 32 x 64
      async_copy16(K + (size_t)(kt + row) * DK_SZ + ch, &Ksh[buf][row][ch]);
    }
    {
      const int row = tid >> 2, ch = (tid & 3) * 8;           // V^T: 64 x 32
      async_copy16(V + (size_t)row * S_LEN + kt + ch, &Vsh[buf][row][ch]);
    }
  };

  const int nt = (qB >> 5) + 4;            // key tiles this block needs
  stage(0, 0);
  wait_async_lds();
  __syncthreads();

  const float sc = 0.125f;                 // 1/sqrt(DK)
  for (int t = 0; t < nt; ++t) {
    const int kt  = t << 5;
    const int cur = t & 1;
    if (t + 1 < nt) stage(cur ^ 1, (t + 1) << 5);  // prefetch next tile

    if (kt <= qw + 15) {                   // wave-uniform causal skip
      const unsigned short* Kb = &Ksh[cur][0][0];
      // K^T B-fragments: key halves t0/t1, dk halves c0/c1 (all LDS b128)
      FragBF kb00, kb01, kb10, kb11;
      const unsigned short* kr0 = Kb + (size_t)ln * DK_SZ;
      const unsigned short* kr1 = Kb + (size_t)(16 + ln) * DK_SZ;
      kb00.q[0] = *(const v4u*)(kr0 + 16 * hi);
      kb00.q[1] = *(const v4u*)(kr0 + 16 * hi + 8);
      kb01.q[0] = *(const v4u*)(kr0 + 32 + 16 * hi);
      kb01.q[1] = *(const v4u*)(kr0 + 32 + 16 * hi + 8);
      kb10.q[0] = *(const v4u*)(kr1 + 16 * hi);
      kb10.q[1] = *(const v4u*)(kr1 + 16 * hi + 8);
      kb11.q[0] = *(const v4u*)(kr1 + 32 + 16 * hi);
      kb11.q[1] = *(const v4u*)(kr1 + 32 + 16 * hi + 8);

      // S = Q @ K^T : two 16x16 score fragments, contraction 64 = 2 WMMA each
      v8f s0 = {}, s1 = {};
      s0 = wmma_bf16(qa0, kb00, s0);
      s0 = wmma_bf16(qa1, kb01, s0);
      s1 = wmma_bf16(qa0, kb10, s1);
      s1 = wmma_bf16(qa1, kb11, s1);

      const bool needmask = (kt + 31 > qw);
      #pragma unroll
      for (int r = 0; r < 8; ++r) {
        float x0 = s0[r] * sc;
        float x1 = s1[r] * sc;
        if (needmask) {                    // causal: key > query -> masked
          const int qrow = qw + r + 8 * hi;
          if (kt + ln > qrow)      x0 = -1.0e30f;
          if (kt + 16 + ln > qrow) x1 = -1.0e30f;
        }
        // online softmax: each row lives across one 16-lane group
        float tm = fmaxf(x0, x1);
        #pragma unroll
        for (int off = 1; off < 16; off <<= 1)
          tm = fmaxf(tm, __shfl_xor(tm, off, 32));
        const float mn = fmaxf(mrow[r], tm);
        const float al = __expf(mrow[r] - mn);
        const float p0 = __expf(x0 - mn);
        const float p1 = __expf(x1 - mn);
        float rs = p0 + p1;
        #pragma unroll
        for (int off = 1; off < 16; off <<= 1)
          rs += __shfl_xor(rs, off, 32);
        lrow[r] = lrow[r] * al + rs;
        mrow[r] = mn;
        alpha[r] = al;
        // stage P (bf16, [16 rows][32 keys]) for A-fragment reload
        pbuf[wave][(r + 8 * hi) * 32 + ln]      = f2bf(p0);
        pbuf[wave][(r + 8 * hi) * 32 + 16 + ln] = f2bf(p1);
      }

      // rescale running context by alpha per row
      #pragma unroll
      for (int nf = 0; nf < 4; ++nf)
        #pragma unroll
        for (int r = 0; r < 8; ++r)
          o[nf][r] *= alpha[r];

      asm volatile("" ::: "memory");  // keep LDS stores before fragment loads
      FragBF pa;                       // P as A-fragment (16x32)
      const unsigned short* pr = &pbuf[wave][ln * 32];
      pa.q[0] = *(const v4u*)(pr + 16 * hi);
      pa.q[1] = *(const v4u*)(pr + 16 * hi + 8);

      // O += P @ V : V^T tile makes each B-fragment 2 contiguous LDS b128
      const unsigned short* Vb = &Vsh[cur][0][0];
      #pragma unroll
      for (int nf = 0; nf < 4; ++nf) {
        FragBF vb;
        const unsigned short* vr = Vb + (size_t)(nf * 16 + ln) * 32 + 16 * hi;
        vb.q[0] = *(const v4u*)(vr);
        vb.q[1] = *(const v4u*)(vr + 8);
        o[nf] = wmma_bf16(pa, vb, o[nf]);
      }
    }

    wait_async_lds();   // next tile's async copies landed
    __syncthreads();    // and everyone is done reading the current buffer
  }

  // normalize and emit ctx bf16 [b][s][h*64+dk]
  float inv[8];
  #pragma unroll
  for (int r = 0; r < 8; ++r) inv[r] = 1.0f / lrow[r];
  #pragma unroll
  for (int nf = 0; nf < 4; ++nf) {
    const int col = h * DK_SZ + nf * 16 + ln;
    #pragma unroll
    for (int r = 0; r < 8; ++r) {
      const int ss = qw + r + 8 * hi;
      ctx[(size_t)(b * S_LEN + ss) * D_DIM + col] = f2bf(o[nf][r] * inv[r]);
    }
  }
}

// ---------------------------------------------------------------------------
extern "C" void kernel_launch(void* const* d_in, const int* in_sizes, int n_in,
                              void* d_out, int out_size, void* d_ws,
                              size_t ws_size, hipStream_t stream) {
  (void)in_sizes; (void)n_in; (void)out_size; (void)ws_size;
  const float* q   = (const float*)d_in[0];
  const float* k   = (const float*)d_in[1];
  const float* v   = (const float*)d_in[2];
  // d_in[3] = causal mask: implemented analytically in flash_attn
  const float* w_q = (const float*)d_in[4];
  const float* b_q = (const float*)d_in[5];
  const float* w_k = (const float*)d_in[6];
  const float* b_k = (const float*)d_in[7];
  const float* w_v = (const float*)d_in[8];
  const float* b_v = (const float*)d_in[9];
  const float* w_o = (const float*)d_in[10];
  const float* b_o = (const float*)d_in[11];

  // workspace: Q, K, Vt, ctx -- each B*H*S*DK bf16 = 8 MiB
  const size_t seg = (size_t)2 * H_NUM * S_LEN * DK_SZ * sizeof(unsigned short);
  char* ws = (char*)d_ws;
  unsigned short* Qh  = (unsigned short*)(ws);
  unsigned short* Kh  = (unsigned short*)(ws + seg);
  unsigned short* Vt  = (unsigned short*)(ws + 2 * seg);
  unsigned short* ctx = (unsigned short*)(ws + 3 * seg);

  const dim3 gg(D_DIM / 64, (2 * S_LEN) / 64);  // (16, 64) tiles
  proj_gemm<0><<<gg, 128, 0, stream>>>(q, nullptr, w_q, b_q, Qh, nullptr);
  proj_gemm<1><<<gg, 128, 0, stream>>>(k, nullptr, w_k, b_k, Kh, nullptr);
  proj_gemm<2><<<gg, 128, 0, stream>>>(v, nullptr, w_v, b_v, Vt, nullptr);
  flash_attn<<<dim3(S_LEN / 128, H_NUM, 2), 256, 0, stream>>>(Qh, Kh, Vt, ctx);
  proj_gemm<3><<<gg, 128, 0, stream>>>(nullptr, ctx, w_o, b_o, nullptr,
                                       (float*)d_out);
}
